// EllEssTeeEmm_Custom_16295105921439
// MI455X (gfx1250) — compile-verified, matrix operand order checked by reading, not run on hardware
//
#include <hip/hip_runtime.h>

typedef __attribute__((ext_vector_type(16))) _Float16 v16h;
typedef __attribute__((ext_vector_type(8)))  float    v8f;

#define BATCH 32
#define LSEQ  1024
#define HDIM  512
#define NWG   64
#define NTHR  256
#define NWAVE 8

__device__ __forceinline__ float sigmoidf_(float x) {
  return 1.0f / (1.0f + __expf(-x));
}

__device__ __forceinline__ unsigned lds_addr32(const void* p) {
  // low 32 bits of a generic pointer to LDS == wave-relative LDS byte offset
  return (unsigned)(unsigned long long)p;
}

__device__ __forceinline__ void async_copy_b128(unsigned lds_off, const void* gsrc) {
  asm volatile("global_load_async_to_lds_b128 %0, %1, off"
               :: "v"(lds_off), "v"((unsigned long long)gsrc)
               : "memory");
}

// prep: zero grid-barrier counters + convert x [B,L,256] f32 -> f16
__global__ void lstm_prep_kernel(const float* __restrict__ x,
                                 unsigned short* __restrict__ x16,
                                 unsigned* __restrict__ sync) {
  if (blockIdx.x == 0 && threadIdx.x < 4) sync[threadIdx.x] = 0u;
  const unsigned n = BATCH * LSEQ * 256;
  for (unsigned e = blockIdx.x * blockDim.x + threadIdx.x; e < n;
       e += gridDim.x * blockDim.x) {
    union { _Float16 f; unsigned short u; } cv;
    cv.f = (_Float16)x[e];
    x16[e] = cv.u;
  }
}

// Persistent LSTM scan for one direction of one layer.
// 64 WGs each own an 8-wide slice of H (32 gate columns). [Whh|Wih] slice is
// f16-resident in LDS; per step the fused GEMM
//   S[32b x 32g] = [h_t | x_t](32 x K) * Wslice^T   (K = 512 + in_sz)
// runs on v_wmma_f32_16x16x32_f16 with K split across 8 waves, partials
// reduced through LDS. Activations are staged with global_load_async_to_lds
// (pure f16 byte moves, ASYNCcnt-tracked). h state double-buffered in global
// f16 => one grid barrier per timestep.
__global__ void __launch_bounds__(NTHR, 1)
lstm_scan_kernel(const unsigned short* __restrict__ X16, int in_sz, int x_layout,
                 const float* __restrict__ Wih, const float* __restrict__ Whh,
                 const float* __restrict__ bih, const float* __restrict__ bhh,
                 unsigned short* __restrict__ h16,   // [2][BATCH][HDIM] f16 bits
                 void* __restrict__ Y, int y_f16, int y_coloff,
                 int reverse, unsigned* __restrict__ sync)
{
  extern __shared__ char smem[];
  const int Ktot  = HDIM + in_sz;
  const int pitch = Ktot + 16;                       // halves; pitch*2 % 16 == 0
  _Float16* A_lds = (_Float16*)smem;                 // [32][pitch]  activations
  _Float16* W_lds = A_lds + 32 * pitch;              // [32][pitch]  weight slice
  float*    P_lds = (float*)(W_lds + 32 * pitch);    // [NWAVE][32][32] partials
  const unsigned A_base = lds_addr32(A_lds);

  const int tid  = threadIdx.x;
  const int wg   = blockIdx.x;
  const int lane = tid & 31;
  const int wave = tid >> 5;
  const int hi   = lane >> 4;
  const int lml  = lane & 15;

  const int b_own = tid & 31;        // this thread's batch row
  const int j_own = tid >> 5;        // this thread's local hidden idx 0..7
  const int jg    = wg * 8 + j_own;  // global hidden index 0..511

  // ---- load weight slice to LDS, f32 -> f16. local col c = gate*8 + jj ----
  for (int e = tid; e < 32 * Ktot; e += NTHR) {
    int c = e / Ktot, k = e - c * Ktot;
    int gate = c >> 3, jj = c & 7;
    int grow = gate * HDIM + wg * 8 + jj;            // row in [4H]
    float w = (k < HDIM) ? Whh[grow * HDIM + k] : Wih[grow * in_sz + (k - HDIM)];
    W_lds[c * pitch + k] = (_Float16)w;
  }

  // biases (NOTE: output-gate uses SCALAR bias[3H] — reference quirk)
  const float bi = bih[jg]            + bhh[jg];
  const float bf = bih[HDIM + jg]     + bhh[HDIM + jg];
  const float bg = bih[2 * HDIM + jg] + bhh[2 * HDIM + jg];
  const float bo = bih[3 * HDIM]      + bhh[3 * HDIM];

  // zero h buffer 0 (each thread owns exactly one (b, jg) cell)
  h16[b_own * HDIM + jg] = 0;
  float c_state = 0.0f;

  unsigned expected = 0;
  auto grid_barrier = [&]() {
    __syncthreads();
    expected += NWG;
    if (tid == 0) {
      __threadfence();                                   // release h writes
      atomicAdd(sync, 1u);
      while (__hip_atomic_load(sync, __ATOMIC_ACQUIRE,
                               __HIP_MEMORY_SCOPE_AGENT) < expected)
        __builtin_amdgcn_s_sleep(1);
    }
    __syncthreads();
  };

  grid_barrier();                       // initial h visible everywhere

  const int Kc  = Ktot / NWAVE;         // 96 (layer0) / 192 (layer1)
  const int k0w = wave * Kc;
  const int xrow_bytes = in_sz * 2;     // f16 row length of x_t
  const int xchunks    = xrow_bytes >> 4;

  for (int s = 0; s < LSEQ; ++s) {
    const int t = reverse ? (LSEQ - 1 - s) : s;
    const unsigned short* hcur = h16 + (s & 1) * (BATCH * HDIM);
    unsigned short*       hnxt = h16 + ((s + 1) & 1) * (BATCH * HDIM);

    // ---- async stage h_t: 32 rows x 128B -> LDS, 1 b128 per thread ----
    {
      int row = tid >> 3, chunk = tid & 7;               // 32 x 8 chunks of 16B
      const char* gsrc = (const char*)hcur + row * (HDIM * 2) + chunk * 16;
      async_copy_b128(A_base + row * (pitch * 2) + chunk * 16, gsrc);
    }
    // ---- async stage x_t: 32 rows x (in_sz*2)B -> LDS after h columns ----
    for (int e = tid; e < 32 * xchunks; e += NTHR) {
      int row = e / xchunks, ch = e - row * xchunks;
      int rowbase = x_layout ? (t * BATCH + row) * in_sz
                             : (row * LSEQ + t) * in_sz;     // halves
      const char* gsrc = (const char*)(X16 + rowbase) + ch * 16;
      async_copy_b128(A_base + row * (pitch * 2) + (HDIM * 2) + ch * 16, gsrc);
    }
    // prefetch next timestep's activations while async copies fly
    {
      int tn = reverse ? (t - 1) : (t + 1);
      if (tn >= 0 && tn < LSEQ) {
        int row = tid & 31, koff = (tid >> 5) * (in_sz >> 3);
        int idx = x_layout ? (tn * BATCH + row) * in_sz + koff
                           : (row * LSEQ + tn) * in_sz + koff;
        __builtin_prefetch(&X16[idx], 0, 1);
      }
    }
    asm volatile("s_wait_asynccnt 0x0" ::: "memory");
    __syncthreads();

    // ---- GEMM over this wave's K-chunk: 2x2 tiles of 16x16 ----
    v8f acc00 = {}, acc01 = {}, acc10 = {}, acc11 = {};
    for (int kb = 0; kb < Kc; kb += 32) {
      const int kbase = k0w + kb;
      v16h a0, a1, b0, b1;
      {
        union { v16h v; unsigned u[8]; } ta, tb;
        #pragma unroll
        for (int r = 0; r < 8; ++r) {
          int kk = kbase + ((r < 4) ? (hi * 8 + 2 * r)
                                    : (16 + hi * 8 + 2 * (r - 4)));
          ta.u[r] = *(const unsigned*)&A_lds[lml * pitch + kk];
          tb.u[r] = *(const unsigned*)&A_lds[(16 + lml) * pitch + kk];
        }
        a0 = ta.v; a1 = tb.v;
      }
      {
        union { v16h v; unsigned u[8]; } ta, tb;
        #pragma unroll
        for (int r = 0; r < 8; ++r) {
          int kk = kbase + ((r < 4) ? (hi * 8 + 2 * r)
                                    : (16 + hi * 8 + 2 * (r - 4)));
          ta.u[r] = *(const unsigned*)&W_lds[lml * pitch + kk];
          tb.u[r] = *(const unsigned*)&W_lds[(16 + lml) * pitch + kk];
        }
        b0 = ta.v; b1 = tb.v;
      }
      acc00 = __builtin_amdgcn_wmma_f32_16x16x32_f16(false, a0, false, b0, (short)0, acc00, false, false);
      acc01 = __builtin_amdgcn_wmma_f32_16x16x32_f16(false, a0, false, b1, (short)0, acc01, false, false);
      acc10 = __builtin_amdgcn_wmma_f32_16x16x32_f16(false, a1, false, b0, (short)0, acc10, false, false);
      acc11 = __builtin_amdgcn_wmma_f32_16x16x32_f16(false, a1, false, b1, (short)0, acc11, false, false);
    }

    // spill partials: C/D layout — VGPR r holds M=r+hi*8, lane%16 = N
    {
      float* P = P_lds + wave * (32 * 32);
      #pragma unroll
      for (int r = 0; r < 8; ++r) {
        int m0 = r + hi * 8;
        P[m0 * 32 + lml]             = acc00[r];
        P[m0 * 32 + 16 + lml]        = acc01[r];
        P[(16 + m0) * 32 + lml]      = acc10[r];
        P[(16 + m0) * 32 + 16 + lml] = acc11[r];
      }
    }
    __syncthreads();

    // ---- reduce K-split partials, gate nonlinearities, state update ----
    float si = 0.f, sf = 0.f, sg = 0.f, so = 0.f;
    #pragma unroll
    for (int w = 0; w < NWAVE; ++w) {
      const float* p = P_lds + (w * 32 + b_own) * 32;
      si += p[j_own];
      sf += p[8 + j_own];
      sg += p[16 + j_own];
      so += p[24 + j_own];
    }
    float iv = sigmoidf_(si + bi);
    float fv = sigmoidf_(sf + bf);
    float gv = tanhf(sg + bg);
    float ov = sigmoidf_(so + bo);
    c_state = fv * c_state + iv * gv;
    float hv = ov * tanhf(c_state);

    union { _Float16 f; unsigned short u; } cv; cv.f = (_Float16)hv;
    hnxt[b_own * HDIM + jg] = cv.u;
    if (y_f16) {   // layer 0 -> x1 f16, [L][B][1024] layout
      ((unsigned short*)Y)[(t * BATCH + b_own) * 1024 + y_coloff + jg] = cv.u;
    } else {       // layer 1 -> out f32, [B][L][1024] layout
      ((float*)Y)[(b_own * LSEQ + t) * 1024 + y_coloff + jg] = hv;
    }

    grid_barrier();     // h_t complete everywhere before next staging
  }
}

extern "C" void kernel_launch(void* const* d_in, const int* in_sizes, int n_in,
                              void* d_out, int out_size, void* d_ws, size_t ws_size,
                              hipStream_t stream) {
  const float* x       = (const float*)d_in[0];
  const float* Wih_f0  = (const float*)d_in[1];
  const float* Whh_f0  = (const float*)d_in[2];
  const float* bih_f0  = (const float*)d_in[3];
  const float* bhh_f0  = (const float*)d_in[4];
  const float* Wih_r0  = (const float*)d_in[5];
  const float* Whh_r0  = (const float*)d_in[6];
  const float* bih_r0  = (const float*)d_in[7];
  const float* bhh_r0  = (const float*)d_in[8];
  const float* Wih_f1  = (const float*)d_in[9];
  const float* Whh_f1  = (const float*)d_in[10];
  const float* bih_f1  = (const float*)d_in[11];
  const float* bhh_f1  = (const float*)d_in[12];
  const float* Wih_r1  = (const float*)d_in[13];
  const float* Whh_r1  = (const float*)d_in[14];
  const float* bih_r1  = (const float*)d_in[15];
  const float* bhh_r1  = (const float*)d_in[16];
  float* out = (float*)d_out;

  char* ws = (char*)d_ws;
  const size_t x16_bytes = (size_t)BATCH * LSEQ * 256 * 2;    // 16 MB
  const size_t x1_bytes  = (size_t)LSEQ * BATCH * 1024 * 2;   // 64 MB
  unsigned short* x16  = (unsigned short*)ws;
  unsigned short* x1   = (unsigned short*)(ws + x16_bytes);
  unsigned short* h16  = (unsigned short*)(ws + x16_bytes + x1_bytes);
  unsigned*       sync = (unsigned*)(ws + x16_bytes + x1_bytes +
                                     2 * BATCH * HDIM * 2);

  lstm_prep_kernel<<<2048, NTHR, 0, stream>>>(x, x16, sync);

  const size_t smem0 = (size_t)2 * 32 * (HDIM + 256 + 16) * 2  + NWAVE * 32 * 32 * 4;
  const size_t smem1 = (size_t)2 * 32 * (HDIM + 1024 + 16) * 2 + NWAVE * 32 * 32 * 4;

  // layer 0, forward:  x16 [B,L,256] -> x1[t][b][0:512]  (f16)
  lstm_scan_kernel<<<NWG, NTHR, smem0, stream>>>(
      x16, 256, 0, Wih_f0, Whh_f0, bih_f0, bhh_f0, h16, x1,  1, 0,   0, sync + 0);
  // layer 0, reverse:  x16 -> x1[t][b][512:1024]  (f16)
  lstm_scan_kernel<<<NWG, NTHR, smem0, stream>>>(
      x16, 256, 0, Wih_r0, Whh_r0, bih_r0, bhh_r0, h16, x1,  1, 512, 1, sync + 1);
  // layer 1, forward:  x1 [L,B,1024] f16 -> out[b][t][0:512]  (f32)
  lstm_scan_kernel<<<NWG, NTHR, smem1, stream>>>(
      x1, 1024, 1, Wih_f1, Whh_f1, bih_f1, bhh_f1, h16, out, 0, 0,   0, sync + 2);
  // layer 1, reverse:  x1 -> out[b][t][512:1024]  (f32)
  lstm_scan_kernel<<<NWG, NTHR, smem1, stream>>>(
      x1, 1024, 1, Wih_r1, Whh_r1, bih_r1, bhh_r1, h16, out, 0, 512, 1, sync + 3);
}
